// sLSTMcell_83227876262153
// MI455X (gfx1250) — matrix operand order repaired and checked by reading, not compiled, not run on hardware
//
#include <hip/hip_runtime.h>
#include <cstdint>

#define D_DIM 2048
#define B_DIM 8
#define T_DIM 512
#define NCOL  8192      // 4 gates * D
#define MROWS 4096      // B * T
#define SCAN_BLOCKS 64

typedef __attribute__((ext_vector_type(16))) __bf16 v16bf;
typedef __attribute__((ext_vector_type(8)))  float  v8f;

union Frag { v16bf v; uint4 q[2]; };

__device__ __forceinline__ unsigned short f32_to_bf16(float f) {
  union { float f; unsigned int u; } c; c.f = f;
  unsigned int u = c.u;
  unsigned int r = (u + 0x7FFFu + ((u >> 16) & 1u)) >> 16;   // round-to-nearest-even
  return (unsigned short)r;
}

// ---------------- utility kernels ----------------

__global__ void cvt_x_kernel(const float* __restrict__ x, unsigned short* __restrict__ xb, int n) {
  for (int i = blockIdx.x * blockDim.x + threadIdx.x; i < n; i += gridDim.x * blockDim.x)
    xb[i] = f32_to_bf16(x[i]);
}

// W_g[e][d] (f32, DxD each) -> dst[d][g*2048 + e] (bf16, 2048 x 8192), LDS-tiled transpose
__global__ void transpose_cvt_kernel(const float* __restrict__ A0, const float* __restrict__ A1,
                                     const float* __restrict__ A2, const float* __restrict__ A3,
                                     unsigned short* __restrict__ dst) {
  __shared__ float tile[32][33];
  const float* src = (blockIdx.z == 0) ? A0 : (blockIdx.z == 1) ? A1 : (blockIdx.z == 2) ? A2 : A3;
  int d0 = blockIdx.x * 32, e0 = blockIdx.y * 32;
  for (int r = threadIdx.y; r < 32; r += 8)
    tile[r][threadIdx.x] = src[(size_t)(e0 + r) * D_DIM + d0 + threadIdx.x];
  __syncthreads();
  for (int r = threadIdx.y; r < 32; r += 8)
    dst[(size_t)(d0 + r) * NCOL + ((size_t)blockIdx.z << 11) + e0 + threadIdx.x] =
        f32_to_bf16(tile[threadIdx.x][r]);
}

__global__ void init_state_kernel(unsigned short* __restrict__ h,
                                  float* __restrict__ c, float* __restrict__ n,
                                  float* __restrict__ m, unsigned int* __restrict__ bar) {
  int i = blockIdx.x * blockDim.x + threadIdx.x;
  if (i < 2 * 16 * D_DIM) h[i] = 0;                       // both h ping/pong buffers, 16 padded rows
  if (i < B_DIM * D_DIM) { c[i] = 0.f; n[i] = 0.f; m[i] = 0.f; }
  if (i == 0) *bar = 0u;                                  // grid barrier counter (reset every call)
}

// ---------------- phase 1: xproj = x @ W^T + b (bf16 WMMA) ----------------

__global__ __launch_bounds__(256) void xproj_gemm_kernel(
    const unsigned short* __restrict__ xb,   // [4096][2048] bf16
    const unsigned short* __restrict__ Wt,   // [2048][8192] bf16 (d-major)
    const float* __restrict__ bz, const float* __restrict__ bi,
    const float* __restrict__ bf_, const float* __restrict__ bo,
    float* __restrict__ xproj) {             // [T][4][B][D] f32
  int wave = threadIdx.x >> 5, lane = threadIdx.x & 31;
  int gw = blockIdx.x * 8 + wave;            // 32768 waves total
  int mtile = gw >> 7;                       // 0..255
  int ntg = gw & 127;                        // 0..127 -> 4 consecutive N tiles
  int col0 = ntg * 64;
  int mrow = lane & 15;
  int kb = (lane < 16) ? 0 : 8;
  const unsigned short* arow = xb + (size_t)(mtile * 16 + mrow) * D_DIM + kb;

  v8f acc0 = {}, acc1 = {}, acc2 = {}, acc3 = {};
  for (int k0 = 0; k0 < D_DIM; k0 += 32) {
    Frag a;
    a.q[0] = *(const uint4*)(arow + k0);
    a.q[1] = *(const uint4*)(arow + k0 + 16);
    Frag b0, b1, b2, b3;
    for (int s = 0; s < 16; ++s) {
      const unsigned short* brow = Wt + (size_t)(k0 + kb + (s & 7) + ((s >> 3) << 4)) * NCOL + col0 + mrow;
      ((unsigned short*)&b0)[s] = brow[0];  ((unsigned short*)&b1)[s] = brow[16];
      ((unsigned short*)&b2)[s] = brow[32]; ((unsigned short*)&b3)[s] = brow[48];
    }
    acc0 = __builtin_amdgcn_wmma_f32_16x16x32_bf16(false, a.v, false, b0.v, (short)0, acc0, false, false);
    acc1 = __builtin_amdgcn_wmma_f32_16x16x32_bf16(false, a.v, false, b1.v, (short)0, acc1, false, false);
    acc2 = __builtin_amdgcn_wmma_f32_16x16x32_bf16(false, a.v, false, b2.v, (short)0, acc2, false, false);
    acc3 = __builtin_amdgcn_wmma_f32_16x16x32_bf16(false, a.v, false, b3.v, (short)0, acc3, false, false);
  }

  int g = col0 >> 11;                        // gate index, constant per wave (2048 % 64 == 0)
  const float* bias = (g == 0) ? bz : (g == 1) ? bi : (g == 2) ? bf_ : bo;
  int nlo = lane & 15;
  int mhi = (lane < 16) ? 0 : 8;
  v8f accs[4] = {acc0, acc1, acc2, acc3};
#pragma unroll
  for (int j = 0; j < 4; ++j) {
    int col = col0 + j * 16 + nlo;
    int e = col & (D_DIM - 1);
    float bval = bias[e];
#pragma unroll
    for (int v = 0; v < 8; ++v) {
      int r = mtile * 16 + v + mhi;          // flattened b*T + t
      int bidx = r >> 9;
      int tt = r & (T_DIM - 1);
      xproj[(((size_t)tt * 4 + g) * B_DIM + bidx) * D_DIM + e] = accs[j][v] + bval;
    }
  }
}

// ---------------- phase 2: persistent recurrent scan ----------------

__device__ __forceinline__ void grid_barrier(unsigned int* bar, unsigned int target) {
  // release: make this block's h/state stores visible at agent scope
  __builtin_amdgcn_fence(__ATOMIC_RELEASE, "agent");
  __syncthreads();
  if (threadIdx.x == 0) {
    __hip_atomic_fetch_add(bar, 1u, __ATOMIC_RELAXED, __HIP_MEMORY_SCOPE_AGENT);
    while (__hip_atomic_load(bar, __ATOMIC_RELAXED, __HIP_MEMORY_SCOPE_AGENT) < target)
      __builtin_amdgcn_s_sleep(2);
  }
  __syncthreads();
  // acquire: invalidate L0 so next step's h reads see other WGPs' writes
  __builtin_amdgcn_fence(__ATOMIC_ACQUIRE, "agent");
}

__global__ __launch_bounds__(256) void scan_persistent_kernel(
    const unsigned short* __restrict__ Rt,    // [2048][8192] bf16
    const float* __restrict__ xproj,          // [T][4][B][D]
    unsigned short* __restrict__ hb,          // 2 x [16][2048] bf16 ping/pong (rows 8..15 zero)
    float* __restrict__ cS, float* __restrict__ nS, float* __restrict__ mS,
    float* __restrict__ out, unsigned int* __restrict__ bar) {
  __shared__ float red[8 * 32 * 8];           // per-wave accumulators for gate combine
  int wave = threadIdx.x >> 5, lane = threadIdx.x & 31;
  int tg = wave >> 2;                         // e-tile group within block (0..1)
  int g  = wave & 3;                          // gate
  int et = blockIdx.x * 2 + tg;               // 0..127
  int mrow = lane & 15;
  int kb = (lane < 16) ? 0 : 8;
  int colbase = (g << 11) + (et << 4);
  const unsigned short* bcol = Rt + colbase;  // + (k0+lane)*NCOL per iter

  for (int t = 0; t < T_DIM; ++t) {
    const unsigned short* hin = hb + (size_t)(t & 1) * 16 * D_DIM;
    unsigned short* hout      = hb + (size_t)((t + 1) & 1) * 16 * D_DIM;
    const unsigned short* arow = hin + mrow * D_DIM + kb;

    v8f acc = {};
    for (int k0 = 0; k0 < D_DIM; k0 += 32) {
      Frag a, b;
      a.q[0] = *(const uint4*)(arow + k0);
      a.q[1] = *(const uint4*)(arow + k0 + 16);
      const unsigned short* bp = bcol + (size_t)(k0 + lane) * NCOL;
      for (int s = 0; s < 16; ++s)
        ((unsigned short*)&b)[s] = bcol[(size_t)(k0 + kb + (s & 7) + ((s >> 3) << 4)) * NCOL + mrow];
      if (k0 + 32 < D_DIM)
        __builtin_prefetch((const void*)(bp + (size_t)32 * NCOL), 0, 3);  // next K-slab of Rt
      acc = __builtin_amdgcn_wmma_f32_16x16x32_bf16(false, a.v, false, b.v, (short)0, acc, false, false);
    }
    *(v8f*)&red[(wave * 32 + lane) * 8] = acc;
    __syncthreads();

    if (g == 0 && lane < 16) {                // lanes 0..15 own N=lane; v=0..7 own batch rows
      int e = (et << 4) + lane;
      size_t xbase = (size_t)t * 4;
#pragma unroll
      for (int v = 0; v < 8; ++v) {
        int b_ = v;
        float pz = red[((tg * 4 + 0) * 32 + lane) * 8 + v] + xproj[((xbase + 0) * B_DIM + b_) * D_DIM + e];
        float pi = red[((tg * 4 + 1) * 32 + lane) * 8 + v] + xproj[((xbase + 1) * B_DIM + b_) * D_DIM + e];
        float pf = red[((tg * 4 + 2) * 32 + lane) * 8 + v] + xproj[((xbase + 2) * B_DIM + b_) * D_DIM + e];
        float po = red[((tg * 4 + 3) * 32 + lane) * 8 + v] + xproj[((xbase + 3) * B_DIM + b_) * D_DIM + e];
        int idx = b_ * D_DIM + e;
        float mp = mS[idx], cp = cS[idx], np = nS[idx];
        float mn = fmaxf(pf + mp, pi);
        float ie = __expf(pi - mn);
        float fe = __expf(pf + mp - mn);
        float z  = tanhf(pz);
        float o  = 1.0f / (1.0f + __expf(-po));
        float cn = fe * cp + ie * z;
        float nn = fe * np + ie;
        float hn = o * (cn / (nn + 1e-8f));
        cS[idx] = cn; nS[idx] = nn; mS[idx] = mn;
        hout[idx] = f32_to_bf16(hn);
        out[((size_t)b_ * T_DIM + t) * D_DIM + e] = hn;
      }
    }

    if (t != T_DIM - 1)
      grid_barrier(bar, (unsigned int)(t + 1) * SCAN_BLOCKS);
  }
}

// ---------------- host launcher ----------------

extern "C" void kernel_launch(void* const* d_in, const int* in_sizes, int n_in,
                              void* d_out, int out_size, void* d_ws, size_t ws_size,
                              hipStream_t stream) {
  const float* x  = (const float*)d_in[0];
  const float* Wz = (const float*)d_in[1];
  const float* Wi = (const float*)d_in[2];
  const float* Wf = (const float*)d_in[3];
  const float* Wo = (const float*)d_in[4];
  const float* bz = (const float*)d_in[5];
  const float* bi = (const float*)d_in[6];
  const float* bf = (const float*)d_in[7];
  const float* bo = (const float*)d_in[8];
  const float* Rz = (const float*)d_in[9];
  const float* Ri = (const float*)d_in[10];
  const float* Rf = (const float*)d_in[11];
  const float* Ro = (const float*)d_in[12];
  float* out = (float*)d_out;

  char* ws = (char*)d_ws;
  constexpr size_t SZ_XB = (size_t)MROWS * D_DIM * 2;        // 16 MB
  constexpr size_t SZ_WT = (size_t)D_DIM * NCOL * 2;         // 32 MB
  constexpr size_t SZ_XP = (size_t)T_DIM * 4 * B_DIM * D_DIM * 4; // 128 MB
  constexpr size_t SZ_H  = (size_t)2 * 16 * D_DIM * 2;       // 128 KB (ping/pong, padded)
  constexpr size_t SZ_S  = (size_t)B_DIM * D_DIM * 4;        // 64 KB each
  size_t off = 0;
  unsigned short* xb = (unsigned short*)(ws + off); off += SZ_XB;
  unsigned short* Wt = (unsigned short*)(ws + off); off += SZ_WT;
  unsigned short* Rt = (unsigned short*)(ws + off); off += SZ_WT;
  float* xproj       = (float*)(ws + off);          off += SZ_XP;
  unsigned short* hb = (unsigned short*)(ws + off); off += SZ_H;
  float* cS          = (float*)(ws + off);          off += SZ_S;
  float* nS          = (float*)(ws + off);          off += SZ_S;
  float* mS          = (float*)(ws + off);          off += SZ_S;
  unsigned int* bar  = (unsigned int*)(ws + off);   off += 256;

  cvt_x_kernel<<<4096, 256, 0, stream>>>(x, xb, MROWS * D_DIM);
  transpose_cvt_kernel<<<dim3(64, 64, 4), dim3(32, 8), 0, stream>>>(Wz, Wi, Wf, Wo, Wt);
  transpose_cvt_kernel<<<dim3(64, 64, 4), dim3(32, 8), 0, stream>>>(Rz, Ri, Rf, Ro, Rt);
  init_state_kernel<<<256, 256, 0, stream>>>(hb, cS, nS, mS, bar);

  xproj_gemm_kernel<<<4096, 256, 0, stream>>>(xb, Wt, bz, bi, bf, bo, xproj);

  scan_persistent_kernel<<<SCAN_BLOCKS, 256, 0, stream>>>(Rt, xproj, hb, cS, nS, mS, out, bar);
}
